// ACELoss_28707561407051
// MI455X (gfx1250) — compile-verified, matrix operand order checked by reading, not run on hardware
//
#include <hip/hip_runtime.h>
#include <hip/hip_bf16.h>
#include <math.h>

typedef float    v4f  __attribute__((ext_vector_type(4)));
typedef float    v8f  __attribute__((ext_vector_type(8)));
typedef _Float16 v16h __attribute__((ext_vector_type(16)));

// ---------------------------------------------------------------------------
// Pass 1: for each row (t,b) of C floats, compute s = max + log(sum exp(x-max))
// One 256-thread block per row. Elements cached in VGPRs (single HBM read),
// non-temporal b128 loads (512MB streamed once; don't pollute 192MB L2).
// ---------------------------------------------------------------------------
template <int VPT>  // C == VPT * 4 * 256
__global__ void ace_pass1_reg(const float* __restrict__ preds,
                              float* __restrict__ s_buf) {
  const int row = blockIdx.x;  // row = t*B + b
  const int tid = threadIdx.x;
  const v4f* rp = (const v4f*)(preds + (size_t)row * (size_t)(VPT * 4 * 256));

  v4f v[VPT];
#pragma unroll
  for (int k = 0; k < VPT; ++k)
    v[k] = __builtin_nontemporal_load(rp + tid + k * 256);

  float m = v[0].x;
#pragma unroll
  for (int k = 0; k < VPT; ++k) {
    m = fmaxf(m, v[k].x); m = fmaxf(m, v[k].y);
    m = fmaxf(m, v[k].z); m = fmaxf(m, v[k].w);
  }

  __shared__ float red[256];
  red[tid] = m;
  __syncthreads();
  for (int off = 128; off > 0; off >>= 1) {
    if (tid < off) red[tid] = fmaxf(red[tid], red[tid + off]);
    __syncthreads();
  }
  m = red[0];
  __syncthreads();

  float z = 0.f;
#pragma unroll
  for (int k = 0; k < VPT; ++k) {
    z += __expf(v[k].x - m); z += __expf(v[k].y - m);
    z += __expf(v[k].z - m); z += __expf(v[k].w - m);
  }
  red[tid] = z;
  __syncthreads();
  for (int off = 128; off > 0; off >>= 1) {
    if (tid < off) red[tid] += red[tid + off];
    __syncthreads();
  }
  if (tid == 0) s_buf[row] = m + __logf(red[0]);
}

// Generic fallback (C not matching a templated size): two streaming reads.
__global__ void ace_pass1_gen(const float* __restrict__ preds,
                              float* __restrict__ s_buf, int C) {
  const int row = blockIdx.x;
  const int tid = threadIdx.x;
  const float* p = preds + (size_t)row * (size_t)C;
  float m = -3.4e38f;
  for (int i = tid; i < C; i += 256) m = fmaxf(m, p[i]);
  __shared__ float red[256];
  red[tid] = m;
  __syncthreads();
  for (int off = 128; off > 0; off >>= 1) {
    if (tid < off) red[tid] = fmaxf(red[tid], red[tid + off]);
    __syncthreads();
  }
  m = red[0];
  __syncthreads();
  float z = 0.f;
  for (int i = tid; i < C; i += 256) z += __expf(p[i] - m);
  red[tid] = z;
  __syncthreads();
  for (int off = 128; off > 0; off >>= 1) {
    if (tid < off) red[tid] += red[tid + off];
    __syncthreads();
  }
  if (tid == 0) s_buf[row] = m + __logf(red[0]);
}

// ---------------------------------------------------------------------------
// Pass 2: one wave (32 lanes) per batch row b.
// For the <=51 needed classes (labels + blank), compute
//   y[c] = sum_t exp(preds[t,b,c] - s[t,b])        (== T * p[b,c])
// via WMMA: D(16x16) = ones(16x32,f16) x E(32x16,f16) + D, accumulated over
// T/32 K-chunks. Row M=0 of D (acc[0], lanes 0..15) carries y for 16 classes.
// partial[b] = sum_j w_j * log(p_j) with w in {1 per positive occurrence,
// T - n_pos for blank}.
// Assumes T == 128 (reference shape); wave32; EXEC all-ones around the WMMA.
// ---------------------------------------------------------------------------
#define ACE_MAX_ENTRIES 128  // >= L+1, multiple of 16

__global__ void ace_pass2_wmma(const float* __restrict__ preds,
                               const int* __restrict__ targets,
                               const float* __restrict__ s_buf,
                               float* __restrict__ partials,
                               int T, int B, int C, int L) {
  const int b    = blockIdx.x;
  const int lane = threadIdx.x;         // 0..31, one wave

  __shared__ float S[128];              // s[t,b] for t = 0..T-1
  __shared__ int   cls[ACE_MAX_ENTRIES];
  __shared__ float wgt[ACE_MAX_ENTRIES];
  __shared__ float red[32];

  for (int t = lane; t < T; t += 32) S[t] = s_buf[(size_t)t * B + b];
  for (int j = lane; j < ACE_MAX_ENTRIES; j += 32) { cls[j] = 0; wgt[j] = 0.f; }
  __syncthreads();

  // Build class list + per-occurrence count weights; count valid positives.
  int npos_local = 0;
  for (int j = lane; j < L; j += 32) {
    int lbl = targets[b * L + j];
    bool valid = ((unsigned)lbl < (unsigned)C);   // out-of-range dropped
    bool pos   = valid && (lbl > 0);
    cls[j] = valid ? lbl : 0;
    wgt[j] = pos ? 1.0f : 0.0f;                   // count per occurrence
    npos_local += pos ? 1 : 0;
  }
  red[lane] = (float)npos_local;
  __syncthreads();
  for (int off = 16; off > 0; off >>= 1) {
    if (lane < off) red[lane] += red[lane + off];
    __syncthreads();
  }
  if (lane == 0) {
    cls[L] = 0;                                    // blank bin
    wgt[L] = (float)T - red[0];                    // count = T - n_pos
  }
  __syncthreads();

  // All-ones f16 A (16x32): layout-independent since every element is 1.
  v16h a;
#pragma unroll
  for (int i = 0; i < 16; ++i) a[i] = (_Float16)1.0f;

  const int laneN  = lane & 15;   // class column N
  const int kh     = lane >> 4;   // which K-half this lane supplies
  const int groups = (L + 1 + 15) / 16;
  const size_t TB_STRIDE = (size_t)B * (size_t)C;  // stride between t planes
  float block_sum = 0.f;

  for (int g = 0; g < groups; ++g) {
    const int c = cls[16 * g + laneN];
    const float* col = preds + (size_t)b * (size_t)C + (size_t)c;

    v8f acc = {};
    for (int cb = 0; cb < T / 32; ++cb) {   // 4 chunks for T=128
      const int tbase = 32 * cb + 16 * kh;  // lanes 0-15: K=0..15; 16-31: K=16..31
      v16h bv;
#pragma unroll
      for (int j = 0; j < 16; ++j) {
        const int t = tbase + j;
        const float e = __expf(col[(size_t)t * TB_STRIDE] - S[t]);
        bv[j] = (_Float16)e;                 // element j <-> K = 16*kh + j
      }
      // D = A(ones) x B(exp) + D  -> every row of D is sum_K B[K][n]
      acc = __builtin_amdgcn_wmma_f32_16x16x32_f16(
          /*neg_a=*/false, a, /*neg_b=*/false, bv,
          /*c_mod=*/(short)0, acc, /*reuse_a=*/false, /*reuse_b=*/false);
    }
    if (lane < 16) {                         // acc[0] = D[0][lane] = y[class]
      const float w = wgt[16 * g + lane];
      if (w != 0.f) {
        const float y = acc[0];              // = T * p[b,c]
        block_sum += w * (__logf(y) - __logf((float)T));
      }
    }
  }

  red[lane] = block_sum;
  __syncthreads();
  for (int off = 16; off > 0; off >>= 1) {
    if (lane < off) red[lane] += red[lane + off];
    __syncthreads();
  }
  if (lane == 0) partials[b] = red[0];       // = sum_j w_j * log(p_j)
}

// ---------------------------------------------------------------------------
// Pass 3: loss = -(1/(B*T)) * sum_b partial[b]  (deterministic, no atomics)
// ---------------------------------------------------------------------------
__global__ void ace_pass3(const float* __restrict__ partials,
                          float* __restrict__ out, int B, int T) {
  __shared__ float red[128];
  const int tid = threadIdx.x;
  float v = 0.f;
  for (int i = tid; i < B; i += 128) v += partials[i];
  red[tid] = v;
  __syncthreads();
  for (int off = 64; off > 0; off >>= 1) {
    if (tid < off) red[tid] += red[tid + off];
    __syncthreads();
  }
  if (tid == 0) out[0] = -red[0] / ((float)B * (float)T);
}

// ---------------------------------------------------------------------------
extern "C" void kernel_launch(void* const* d_in, const int* in_sizes, int n_in,
                              void* d_out, int out_size, void* d_ws, size_t ws_size,
                              hipStream_t stream) {
  const float* preds   = (const float*)d_in[0];
  const int*   targets = (const int*)d_in[1];
  float*       out     = (float*)d_out;

  const int T = 128, B = 128;                       // reference shapes
  const int C = in_sizes[0] / (T * B);              // 8192
  const int L = in_sizes[1] / B;                    // 50

  float* s_buf    = (float*)d_ws;                   // T*B floats
  float* partials = s_buf + (size_t)T * (size_t)B;  // B floats

  const int rows = T * B;
  if (C == 8192)      ace_pass1_reg<8><<<rows, 256, 0, stream>>>(preds, s_buf);
  else if (C == 4096) ace_pass1_reg<4><<<rows, 256, 0, stream>>>(preds, s_buf);
  else if (C == 2048) ace_pass1_reg<2><<<rows, 256, 0, stream>>>(preds, s_buf);
  else if (C == 1024) ace_pass1_reg<1><<<rows, 256, 0, stream>>>(preds, s_buf);
  else                ace_pass1_gen<<<rows, 256, 0, stream>>>(preds, s_buf, C);

  ace_pass2_wmma<<<B, 32, 0, stream>>>(preds, targets, s_buf, partials, T, B, C, L);
  ace_pass3<<<1, 128, 0, stream>>>(partials, out, B, T);
}
